// PatchBatchRelationalModule_75935021793485
// MI455X (gfx1250) — matrix operand order repaired and checked by reading, not compiled, not run on hardware
//
#include <hip/hip_runtime.h>
#include <hip/hip_bf16.h>

// ---------------------------------------------------------------------------
// PatchBatchRelationalModule on MI455X (gfx1250, wave32, WMMA bf16)
//
//   pair@W0 = feats_j @ W0[:130] + feats_i @ W0[130:]
//   Arow[b,j,:] = feats[b,j] @ W0_top               (tiny f32 GEMM)
//   Brow[b,i,:] = feats[b,i] @ W0_bot + b0
//   H0[q=(i,j)] = leaky(Arow[j] + Brow[i])          (built on the fly, bf16)
//   H1          = leaky(H0 @ W1 + b1)               (v_wmma_f32_16x16x32_bf16)
//   out[b]      = (mean_q H1[q]) @ Wout + bout
//
// Main-loop nest: n-tiles processed in groups of 4 with k outermost inside a
// group, so each A-fragment pair (16 VGPRs) feeds 8 WMMAs and the live set
// (4x2 accumulators = 64 VGPRs + small A/B windows) stays well under 256
// VGPRs -> no s_set_vgpr_msb churn, ~1.5 ds_load_b128 per v_wmma.
// ---------------------------------------------------------------------------

typedef __attribute__((ext_vector_type(16))) __bf16 v16bf;
typedef __attribute__((ext_vector_type(8)))  __bf16 v8bf;
typedef __attribute__((ext_vector_type(2)))  __bf16 v2bf;
typedef __attribute__((ext_vector_type(8)))  float  v8f;

#define PGRID   15
#define NP      225               // nodes per batch
#define PP      (NP * NP)         // 50625 pairs per batch
#define DFEAT   130
#define HID     256
#define NOUTD   128
#define NBATCH  8
#define TROWS   32                // rows (pairs) per wave tile
#define NT32    ((PP + TROWS - 1) / TROWS)  // 1583 tiles (last has 1 valid row)
#define H0PITCH 264               // 256 + 8 bf16 pad -> break LDS bank conflicts

// workspace layout (floats)
#define WS_AROW 0
#define WS_BROW (NBATCH * NP * HID)
#define WS_SUM  (2 * NBATCH * NP * HID)

__global__ void zero_sums_kernel(float* __restrict__ sums) {
    int i = blockIdx.x * blockDim.x + threadIdx.x;
    if (i < NBATCH * HID) sums[i] = 0.0f;
}

// One block per (b, p): pool x -> 130 features in LDS, then each of the 256
// threads computes one column of Arow and Brow (260 coalesced W0 reads each).
__global__ __launch_bounds__(256)
void proj_kernel(const float* __restrict__ x, const float* __restrict__ W0,
                 const float* __restrict__ b0,
                 float* __restrict__ Arow, float* __restrict__ Brow) {
    __shared__ float feat[DFEAT];
    const int bp = blockIdx.x;            // b*225 + p
    const int b  = bp / NP;
    const int p  = bp - b * NP;
    const int pr = p / PGRID, pc = p - pr * PGRID;
    const int t  = threadIdx.x;

    if (t < DFEAT) {
        float v;
        if (t < 128) {
            const int c  = t >> 2;
            const int kr = (t >> 1) & 1, kc = t & 1;
            const int y  = pr + kr, xx = pc + kc;    // pooled 16x16 coords
            const float* px = x + (((size_t)(b * 32 + c) * 32 + 2 * y) * 32 + 2 * xx);
            v = 0.25f * (px[0] + px[1] + px[32] + px[33]);
        } else if (t == 128) {
            v = (float)(pc - 7);                      // cx
        } else {
            v = (float)(pr - 7);                      // cy
        }
        feat[t] = v;
    }
    __syncthreads();

    const int n = t;                                  // 0..255
    float accA = 0.0f, accB = b0[n];
    for (int d = 0; d < DFEAT; ++d) {
        const float f = feat[d];
        accA += f * W0[d * HID + n];                  // W0_top
        accB += f * W0[(DFEAT + d) * HID + n];        // W0_bot
    }
    Arow[(size_t)bp * HID + n] = accA;
    Brow[(size_t)bp * HID + n] = accB;
}

__device__ __forceinline__ float leaky(float h) {
    return fmaxf(h, 0.0f) + 0.01f * fminf(h, 0.0f);
}

__device__ __forceinline__ v16bf make_afrag(const __bf16* pa) {
    // 16-bit A-matrix 16x32 layout: a lane's 16 bf16 = K{off..off+7, off+16..off+23}
    const v8bf lo = *(const v8bf*)pa;
    const v8bf hc = *(const v8bf*)(pa + 16);
    v16bf a;
#pragma unroll
    for (int e = 0; e < 8; ++e) { a[e] = lo[e]; a[e + 8] = hc[e]; }
    return a;
}

// Main kernel: 128 threads = 4 waves. W1 staged once per block into LDS in
// exact 16x16x32 bf16 B-fragment order; each wave streams 32-row H0 tiles,
// 256 WMMAs per tile in 4 n-tile groups with k-outer A-fragment reuse.
extern "C" __global__ __launch_bounds__(128)
void relpair_kernel(const float* __restrict__ Arow, const float* __restrict__ Brow,
                    const float* __restrict__ W1,   const float* __restrict__ b1,
                    float* __restrict__ sums) {
    extern __shared__ char smem[];
    __bf16* w1f = (__bf16*)smem;                              // 256*256 bf16 = 128 KB
    __bf16* h0  = (__bf16*)(smem + HID * HID * 2);            // 4 * TROWS * H0PITCH bf16

    const int tid    = threadIdx.x;
    const int wave   = tid >> 5;
    const int lane   = tid & 31;
    const int lanelo = lane & 15;
    const int hi16   = lane >> 4;
    const int b      = blockIdx.x;

    // --- stage W1 (f32 -> bf16) into LDS in B-fragment order ---------------
    // B 32x16 layout per (t,k): lanes 0-15 hold K {0..7,16..23} of column N=lane,
    // lanes 16-31 hold K {8..15,24..31} of column N=lane-16.
    for (int idx = tid; idx < HID * HID; idx += blockDim.x) {
        const int n     = idx & 255;           // coalesced global read: n fastest
        const int kk    = idx >> 8;
        const int t8    = n >> 4;              // n tile
        const int nl    = n & 15;
        const int k8    = kk >> 5;             // k tile
        const int rem   = kk & 31;
        const int lhalf = (rem >> 3) & 1;
        const int e     = (rem & 7) + ((rem >> 4) << 3);
        const int dst   = (((t8 * 8 + k8) * 32) + (lhalf * 16 + nl)) * 16 + e;
        w1f[dst] = (__bf16)W1[idx];
    }
    __syncthreads();

    // per-lane bias for each n tile: column n = t*16 + lanelo
    float biasv[16];
#pragma unroll
    for (int t = 0; t < 16; ++t) biasv[t] = b1[t * 16 + lanelo];

    float colsum[16];
#pragma unroll
    for (int t = 0; t < 16; ++t) colsum[t] = 0.0f;

    __bf16* h0w = h0 + wave * TROWS * H0PITCH;
    const float* Ab = Arow + (size_t)b * NP * HID;
    const float* Bb = Brow + (size_t)b * NP * HID;

    for (int tile = blockIdx.y * 4 + wave; tile < NT32; tile += gridDim.y * 4) {
        const int q0 = tile * TROWS;

        // --- build 32x256 H0 tile (leaky(A[j]+B[i])) as packed bf16 --------
        for (int r = 0; r < TROWS; ++r) {
            int q = q0 + r;
            if (q >= PP) q = PP - 1;           // masked later
            const int i = q / NP;
            const int j = q - i * NP;
            const float2* Aj = (const float2*)(Ab + (size_t)j * HID);
            const float2* Bi = (const float2*)(Bb + (size_t)i * HID);
#pragma unroll
            for (int cp = lane; cp < HID / 2; cp += 32) {   // column pair
                const float2 av = Aj[cp];
                const float2 bv = Bi[cp];
                v2bf pk;
                pk[0] = (__bf16)leaky(av.x + bv.x);
                pk[1] = (__bf16)leaky(av.y + bv.y);
                *(v2bf*)&h0w[r * H0PITCH + cp * 2] = pk;
            }
        }
        asm volatile("s_wait_dscnt 0" ::: "memory");

        // A-fragment bases for the two 16-row subtiles.
        // lane<16: row M=lane, K offsets {0..7,16..23};
        // lane>=16: row M=lane-16, K offsets {8..15,24..31}.
        const __bf16* ab0 = h0w + lanelo * H0PITCH + hi16 * 8;
        const __bf16* ab1 = ab0 + 16 * H0PITCH;

        const bool fulltile = (q0 + TROWS) <= PP;

        // --- 4 n-tile groups; k outermost inside a group so each A-fragment
        //     pair feeds 4 n-tiles x 2 subtiles = 8 WMMAs ---------------------
#pragma unroll
        for (int tg = 0; tg < 4; ++tg) {
            v8f acc0[4] = {{}, {}, {}, {}};
            v8f acc1[4] = {{}, {}, {}, {}};
#pragma unroll
            for (int k = 0; k < 8; ++k) {
                const v16bf A0 = make_afrag(ab0 + k * 32);
                const v16bf A1 = make_afrag(ab1 + k * 32);
#pragma unroll
                for (int tt = 0; tt < 4; ++tt) {
                    const int t = tg * 4 + tt;
                    const v16bf bfrag =
                        *(const v16bf*)&w1f[(((t * 8 + k) * 32) + lane) * 16];
                    acc0[tt] = __builtin_amdgcn_wmma_f32_16x16x32_bf16(
                            false, A0, false, bfrag, (short)0, acc0[tt], false, false);
                    acc1[tt] = __builtin_amdgcn_wmma_f32_16x16x32_bf16(
                            false, A1, false, bfrag, (short)0, acc1[tt], false, false);
                }
            }
            // C/D layout: VGPR r holds (M = hi16*8 + r, N = t*16 + lanelo)
            if (fulltile) {
#pragma unroll
                for (int tt = 0; tt < 4; ++tt) {
                    const int t = tg * 4 + tt;
                    const float bias = biasv[t];
                    float s = 0.0f;
#pragma unroll
                    for (int r = 0; r < 8; ++r)
                        s += leaky(acc0[tt][r] + bias) + leaky(acc1[tt][r] + bias);
                    colsum[t] += s;
                }
            } else {
#pragma unroll
                for (int tt = 0; tt < 4; ++tt) {
                    const int t = tg * 4 + tt;
                    const float bias = biasv[t];
                    float s = 0.0f;
#pragma unroll
                    for (int r = 0; r < 8; ++r) {
                        const int M = hi16 * 8 + r;
                        s += ((q0 + M)      < PP) ? leaky(acc0[tt][r] + bias) : 0.0f;
                        s += ((q0 + 16 + M) < PP) ? leaky(acc1[tt][r] + bias) : 0.0f;
                    }
                    colsum[t] += s;
                }
            }
        }
    }

    float* sb = sums + b * HID;
#pragma unroll
    for (int t = 0; t < 16; ++t)
        atomicAdd(&sb[t * 16 + lanelo], colsum[t]);
}

// out[b,m] = (sum_n sums[b,n]/50625) * Wout[n,m] + bout[m]
__global__ __launch_bounds__(128)
void final_kernel(const float* __restrict__ sums, const float* __restrict__ Wout,
                  const float* __restrict__ bout, float* __restrict__ out) {
    const int b = blockIdx.x;
    const int m = threadIdx.x;                 // 0..127
    const float inv = 1.0f / (float)PP;
    float acc = bout[m];
    const float* s = sums + b * HID;
    for (int n = 0; n < HID; ++n)
        acc += (s[n] * inv) * Wout[n * NOUTD + m];
    out[b * NOUTD + m] = acc;
}

extern "C" void kernel_launch(void* const* d_in, const int* in_sizes, int n_in,
                              void* d_out, int out_size, void* d_ws, size_t ws_size,
                              hipStream_t stream) {
    const float* x    = (const float*)d_in[0];
    const float* W0   = (const float*)d_in[1];
    const float* b0   = (const float*)d_in[2];
    const float* W1   = (const float*)d_in[3];
    const float* b1   = (const float*)d_in[4];
    const float* Wout = (const float*)d_in[5];
    const float* bout = (const float*)d_in[6];

    float* ws   = (float*)d_ws;
    float* Arow = ws + WS_AROW;
    float* Brow = ws + WS_BROW;
    float* sums = ws + WS_SUM;
    float* out  = (float*)d_out;

    const size_t shmem = (size_t)HID * HID * 2 + (size_t)4 * TROWS * H0PITCH * 2; // ~194 KB
    hipFuncSetAttribute((const void*)relpair_kernel,
                        hipFuncAttributeMaxDynamicSharedMemorySize, (int)shmem);

    zero_sums_kernel<<<dim3((NBATCH * HID + 255) / 256), dim3(256), 0, stream>>>(sums);
    proj_kernel<<<dim3(NBATCH * NP), dim3(256), 0, stream>>>(x, W0, b0, Arow, Brow);
    relpair_kernel<<<dim3(NBATCH, 40), dim3(128), shmem, stream>>>(Arow, Brow, W1, b1, sums);
    final_kernel<<<dim3(NBATCH), dim3(128), 0, stream>>>(sums, Wout, bout, out);
}
